// Model_88072599372209
// MI455X (gfx1250) — compile-verified, matrix operand order checked by reading, not run on hardware
//
#include <hip/hip_runtime.h>
#include <hip/hip_bf16.h>

typedef __attribute__((ext_vector_type(16))) _Float16 v16h;
typedef __attribute__((ext_vector_type(8)))  float    v8f;

// ---------------------------------------------------------------------------
// WMMA helpers
// ---------------------------------------------------------------------------
__device__ __forceinline__ v8f wmma_f16(v16h a, v16h b, v8f c) {
    // D = A(16x32 f16) * B(32x16 f16) + C(16x16 f32)
    return __builtin_amdgcn_wmma_f32_16x16x32_f16(false, a, false, b, (short)0, c,
                                                  false, false);
}

// Load a 16x32 f16 fragment (A-layout) from an LDS tile stored row-major with
// K contiguous.  Per CDNA5 ISA 16-bit A-matrix layout:
//   lane<16 : M=lane,    K halves {0..7} then {16..23} -> dwords {0..3},{8..11}
//   lane>=16: M=lane-16, K halves {8..15} then {24..31}-> dwords {4..7},{12..15}
__device__ __forceinline__ v16h lds_frag(const _Float16* base, int rowBase, int lane,
                                         int strideH, int colOffH) {
    const uint32_t* p =
        (const uint32_t*)(base + (size_t)(rowBase + (lane & 15)) * strideH + colOffH);
    const int o = (lane >> 4) * 4;
    union { uint32_t u[8]; v16h h; } r;
#pragma unroll
    for (int i = 0; i < 4; i++) r.u[i] = p[o + i];
#pragma unroll
    for (int i = 0; i < 4; i++) r.u[4 + i] = p[o + 8 + i];
    return r.h;
}

// ---------------------------------------------------------------------------
// Async global->LDS copy (CDNA5 GLOBAL_LOAD_ASYNC_TO_LDS_B128, ASYNCcnt).
// lds_off = LDS byte address (low 32 bits of the generic shared address;
// flat->LDS mapping truncates to addr[31:0]).  16 bytes per active lane.
// ---------------------------------------------------------------------------
__device__ __forceinline__ void async_copy16(uint32_t lds_off, const void* gptr) {
#if defined(__HIP_DEVICE_COMPILE__)
    asm volatile("global_load_async_to_lds_b128 %0, %1, off"
                 :: "v"(lds_off), "v"(gptr)
                 : "memory");
#else
    (void)lds_off; (void)gptr;
#endif
}

__device__ __forceinline__ void wait_async0() {
#if defined(__HIP_DEVICE_COMPILE__)
    asm volatile("s_wait_asynccnt 0x0" ::: "memory");
#endif
}

__device__ __forceinline__ uint32_t lds_addr32(const void* p) {
    return (uint32_t)(uintptr_t)p;
}

// ---------------------------------------------------------------------------
// Generic WMMA GEMM:  C[M,N] = A[M,K] * B[K,N]      (A,B f32 in memory,
// converted to f16 on the way into LDS; f32 accumulation in WMMA)
// OUTMODE: 0 -> C = acc (f32);  1 -> C += acc (f32);  2 -> C = (f16)acc
// Requires M%128==0, N%128==0, K%32==0.
// Block: 256 threads (8 wave32 waves, 4 along M x 2 along N, 32x64 per wave).
// ---------------------------------------------------------------------------
template <int OUTMODE>
__global__ void __launch_bounds__(256)
gemm_f16wmma(const float* __restrict__ A, const float* __restrict__ B,
             void* __restrict__ Cv, int M, int N, int K) {
    __shared__ __align__(16) _Float16 As[128][40];
    __shared__ __align__(16) _Float16 Bt[128][40];   // transposed: [n][k]

    const int tid  = threadIdx.x;
    const int lane = tid & 31;
    const int wid  = tid >> 5;
    const int wm   = (wid & 3) * 32;    // wave M offset in block tile
    const int wn   = (wid >> 2) * 64;   // wave N offset in block tile
    const int mBase = blockIdx.y * 128;
    const int nBase = blockIdx.x * 128;

    const v8f zf = {0, 0, 0, 0, 0, 0, 0, 0};
    v8f acc[2][4];
#pragma unroll
    for (int i = 0; i < 2; i++)
#pragma unroll
        for (int j = 0; j < 4; j++) acc[i][j] = zf;

    for (int kb = 0; kb < K; kb += 32) {
        __syncthreads();   // previous compute done before overwriting tiles
        // ---- A tile: 128x32 f32 -> f16 LDS (row-major, K contiguous)
#pragma unroll
        for (int i = 0; i < 4; i++) {
            int u = tid + i * 256;
            int r = u >> 3, c4 = u & 7;
            const float4 av =
                *reinterpret_cast<const float4*>(A + (size_t)(mBase + r) * K + kb + c4 * 4);
            As[r][c4 * 4 + 0] = (_Float16)av.x;
            As[r][c4 * 4 + 1] = (_Float16)av.y;
            As[r][c4 * 4 + 2] = (_Float16)av.z;
            As[r][c4 * 4 + 3] = (_Float16)av.w;
        }
        // ---- B tile: 32x128 f32 -> transposed f16 LDS Bt[n][k]
#pragma unroll
        for (int i = 0; i < 4; i++) {
            int u = tid + i * 256;
            int r = u >> 5, c4 = u & 31;
            const float4 bv =
                *reinterpret_cast<const float4*>(B + (size_t)(kb + r) * N + nBase + c4 * 4);
            Bt[c4 * 4 + 0][r] = (_Float16)bv.x;
            Bt[c4 * 4 + 1][r] = (_Float16)bv.y;
            Bt[c4 * 4 + 2][r] = (_Float16)bv.z;
            Bt[c4 * 4 + 3][r] = (_Float16)bv.w;
        }
        if (kb + 32 < K) {   // hint next tiles toward the caches
            __builtin_prefetch(A + (size_t)(mBase + (tid >> 3)) * K + kb + 32, 0, 1);
            __builtin_prefetch(B + (size_t)(kb + 32 + (tid >> 5)) * N + nBase + (tid & 31) * 4, 0, 1);
        }
        __syncthreads();

        v16h aF[2], bF[4];
#pragma unroll
        for (int mt = 0; mt < 2; mt++) aF[mt] = lds_frag(&As[0][0], wm + mt * 16, lane, 40, 0);
#pragma unroll
        for (int nt = 0; nt < 4; nt++) bF[nt] = lds_frag(&Bt[0][0], wn + nt * 16, lane, 40, 0);
#pragma unroll
        for (int mt = 0; mt < 2; mt++)
#pragma unroll
            for (int nt = 0; nt < 4; nt++)
                acc[mt][nt] = wmma_f16(aF[mt], bF[nt], acc[mt][nt]);
    }

    // ---- epilogue: C layout VGPR r -> row r (lanes<16) / r+8 (lanes>=16)
    const int rOff = (lane >> 4) * 8;
    const int cOff = lane & 15;
#pragma unroll
    for (int mt = 0; mt < 2; mt++)
#pragma unroll
        for (int nt = 0; nt < 4; nt++)
#pragma unroll
            for (int r = 0; r < 8; r++) {
                size_t row = (size_t)(mBase + wm + mt * 16 + rOff + r);
                size_t col = (size_t)(nBase + wn + nt * 16 + cOff);
                float vv = acc[mt][nt][r];
                if (OUTMODE == 2)
                    ((_Float16*)Cv)[row * N + col] = (_Float16)vv;
                else if (OUTMODE == 1)
                    ((float*)Cv)[row * N + col] += vv;
                else
                    ((float*)Cv)[row * N + col] = vv;
            }
}

// ---------------------------------------------------------------------------
// Fused flash attention.  grid = (L/64, NH, B), block = 128 (4 waves).
// Wave w owns 16 query rows; online softmax; O accumulated in f32.
// q/k/v: f16 [B, L, E] with head h occupying cols h*64..h*64+63.
// out: f32 [B, L, E].
// Q and K tiles are moved with CDNA5 async global->LDS transfers.
// ---------------------------------------------------------------------------
__global__ void __launch_bounds__(128)
flash_attn(const _Float16* __restrict__ qp, const _Float16* __restrict__ kp,
           const _Float16* __restrict__ vp, const int* __restrict__ maskp,
           float* __restrict__ outp) {
    constexpr int L = 1024, E = 512;
    __shared__ __align__(16) _Float16 Qs[64][72];     // 144 B/row (16B aligned)
    __shared__ __align__(16) _Float16 Ks[32][72];
    __shared__ __align__(16) _Float16 Vt[64][40];     // [dim][key]
    __shared__ __align__(16) _Float16 Ps[4][16][40];  // per-wave P staging

    const int tid = threadIdx.x, lane = tid & 31, w = tid >> 5;
    const int qb = blockIdx.x, hh = blockIdx.y, b = blockIdx.z;
    const int q0 = qb * 64;
    const float scale = 0.125f;  // 1/sqrt(64)

    // ---- async copy of the 64x64 Q tile into LDS (16 B per lane-transfer)
    {
        const uint32_t qs_base = lds_addr32(&Qs[0][0]);
        const _Float16* qg = qp + ((size_t)b * L + q0) * E + hh * 64;
        for (int i = tid; i < 64 * 8; i += 128) {
            int r = i >> 3, c = i & 7;
            async_copy16(qs_base + (uint32_t)(r * 144 + c * 16), qg + (size_t)r * E + c * 8);
        }
        wait_async0();
    }

    const v8f zf = {0, 0, 0, 0, 0, 0, 0, 0};
    float mrow[8], lrow[8];
#pragma unroll
    for (int r = 0; r < 8; r++) { mrow[r] = -3.0e38f; lrow[r] = 0.f; }
    v8f oacc[4];
#pragma unroll
    for (int t = 0; t < 4; t++) oacc[t] = zf;

    const int rOff = (lane >> 4) * 8;
    const int cIdx = lane & 15;
    const uint32_t ks_base = lds_addr32(&Ks[0][0]);

    for (int j0 = 0; j0 < L; j0 += 32) {
        __syncthreads();  // previous iteration's reads of Ks/Vt done
        // ---- K tile: async global->LDS (straight copy, B-transposed layout)
        {
            const _Float16* kg = kp + ((size_t)b * L + j0) * E + hh * 64;
            for (int i = tid; i < 32 * 8; i += 128) {
                int r = i >> 3, c = i & 7;
                async_copy16(ks_base + (uint32_t)(r * 144 + c * 16), kg + (size_t)r * E + c * 8);
            }
        }
        // ---- V tile: VALU transpose into Vt[dim][key]
        {
            const uint32_t* vg = (const uint32_t*)(vp + ((size_t)b * L + j0) * E + hh * 64);
            for (int i = tid; i < 32 * 32; i += 128) {
                int r = i >> 5, c = i & 31;
                union { uint32_t u; _Float16 f[2]; } dv;
                dv.u = vg[(size_t)r * 256 + c];
                Vt[c * 2 + 0][r] = dv.f[0];
                Vt[c * 2 + 1][r] = dv.f[1];
            }
        }
        wait_async0();
        __syncthreads();

        // ---- S = Q * K^T  (rows of Ks are already in B-transposed layout)
        v16h aQ0 = lds_frag(&Qs[0][0], w * 16, lane, 72, 0);
        v16h aQ1 = lds_frag(&Qs[0][0], w * 16, lane, 72, 32);
        v8f s0 = zf, s1 = zf;
        {
            v16h b00 = lds_frag(&Ks[0][0], 0, lane, 72, 0);
            v16h b01 = lds_frag(&Ks[0][0], 0, lane, 72, 32);
            v16h b10 = lds_frag(&Ks[0][0], 16, lane, 72, 0);
            v16h b11 = lds_frag(&Ks[0][0], 16, lane, 72, 32);
            s0 = wmma_f16(aQ0, b00, s0);
            s0 = wmma_f16(aQ1, b01, s0);
            s1 = wmma_f16(aQ0, b10, s1);
            s1 = wmma_f16(aQ1, b11, s1);
        }
        const float am0 = maskp[(size_t)b * L + j0 + cIdx] ? 0.f : -1e9f;
        const float am1 = maskp[(size_t)b * L + j0 + 16 + cIdx] ? 0.f : -1e9f;

        // ---- online softmax (row stats reduced across each 16-lane half)
        float p0[8], p1[8], alpha[8];
#pragma unroll
        for (int r = 0; r < 8; r++) {
            p0[r] = s0[r] * scale + am0;
            p1[r] = s1[r] * scale + am1;
            float mc = fmaxf(p0[r], p1[r]);
            mc = fmaxf(mc, __shfl_xor(mc, 1, 16));
            mc = fmaxf(mc, __shfl_xor(mc, 2, 16));
            mc = fmaxf(mc, __shfl_xor(mc, 4, 16));
            mc = fmaxf(mc, __shfl_xor(mc, 8, 16));
            float mnew = fmaxf(mrow[r], mc);
            alpha[r] = __expf(mrow[r] - mnew);
            p0[r] = __expf(p0[r] - mnew);
            p1[r] = __expf(p1[r] - mnew);
            float rs = p0[r] + p1[r];
            rs += __shfl_xor(rs, 1, 16);
            rs += __shfl_xor(rs, 2, 16);
            rs += __shfl_xor(rs, 4, 16);
            rs += __shfl_xor(rs, 8, 16);
            lrow[r] = lrow[r] * alpha[r] + rs;
            mrow[r] = mnew;
        }
#pragma unroll
        for (int t = 0; t < 4; t++)
#pragma unroll
            for (int r = 0; r < 8; r++) oacc[t][r] *= alpha[r];

        // ---- P (C-layout) -> LDS -> A-layout fragment; O += P * V
#pragma unroll
        for (int r = 0; r < 8; r++) {
            Ps[w][rOff + r][cIdx]      = (_Float16)p0[r];
            Ps[w][rOff + r][cIdx + 16] = (_Float16)p1[r];
        }
        v16h pF = lds_frag(&Ps[w][0][0], 0, lane, 40, 0);
#pragma unroll
        for (int t = 0; t < 4; t++) {
            v16h bV = lds_frag(&Vt[0][0], t * 16, lane, 40, 0);
            oacc[t] = wmma_f16(pF, bV, oacc[t]);
        }
    }

    // ---- normalize and write
    float rinv[8];
#pragma unroll
    for (int r = 0; r < 8; r++) rinv[r] = 1.0f / lrow[r];
#pragma unroll
    for (int t = 0; t < 4; t++)
#pragma unroll
        for (int r = 0; r < 8; r++) {
            size_t row = (size_t)b * L + q0 + w * 16 + rOff + r;
            outp[row * E + hh * 64 + t * 16 + cIdx] = oacc[t][r] * rinv[r];
        }
}

// ---------------------------------------------------------------------------
// LayerNorm over E=512; one block (256 threads) per row.
// ---------------------------------------------------------------------------
__global__ void __launch_bounds__(256)
ln_kernel(const float* __restrict__ hp, const float* __restrict__ g,
          const float* __restrict__ bta, float* __restrict__ outp) {
    const int row = blockIdx.x;
    const int tid = threadIdx.x;
    const float x0 = hp[(size_t)row * 512 + tid];
    const float x1 = hp[(size_t)row * 512 + tid + 256];
    float s = x0 + x1;
    float q = x0 * x0 + x1 * x1;
#pragma unroll
    for (int d = 16; d >= 1; d >>= 1) {
        s += __shfl_xor(s, d, 32);
        q += __shfl_xor(q, d, 32);
    }
    __shared__ float red[16];
    __shared__ float stats[2];
    const int lane = tid & 31, wid = tid >> 5;
    if (lane == 0) { red[wid] = s; red[8 + wid] = q; }
    __syncthreads();
    if (tid == 0) {
        float ts = 0.f, tq = 0.f;
        for (int i = 0; i < 8; i++) { ts += red[i]; tq += red[8 + i]; }
        float mean = ts * (1.0f / 512.0f);
        float var  = tq * (1.0f / 512.0f) - mean * mean;
        stats[0] = mean;
        stats[1] = rsqrtf(var + 1e-5f);
    }
    __syncthreads();
    const float mean = stats[0], rstd = stats[1];
    outp[(size_t)row * 512 + tid]       = (x0 - mean) * rstd * g[tid] + bta[tid];
    outp[(size_t)row * 512 + tid + 256] = (x1 - mean) * rstd * g[tid + 256] + bta[tid + 256];
}

// ---------------------------------------------------------------------------
// Input projection (K=16) + cosine 2D positional embedding.
// ---------------------------------------------------------------------------
__global__ void __launch_bounds__(256)
input_proj(const float* __restrict__ x, const float* __restrict__ Win,
           const float* __restrict__ bin, float* __restrict__ hp) {
    const int idx = blockIdx.x * 256 + threadIdx.x;  // < 8192*512
    const int m = idx >> 9, e = idx & 511;
    const float* xr = x + (size_t)m * 16;
    float acc = bin[e];
#pragma unroll
    for (int k2 = 0; k2 < 16; k2++) acc += xr[k2] * Win[k2 * 512 + e];
    const int l  = m & 1023;             // token position (L=1024)
    const int ph = l >> 5, pw = l & 31;  // H=W=32
    const int fi    = (e < 256) ? e : (e - 256);
    const float pos = (e < 256) ? (float)ph : (float)pw;
    const float inv = __expf((float)fi * (-9.210340371976184f / 256.0f));
    acc += __cosf(pos * inv);
    hp[idx] = acc;
}

// ---------------------------------------------------------------------------
// a = silu(a) * g  (elementwise over 8192*2048)
// ---------------------------------------------------------------------------
__global__ void __launch_bounds__(256)
silu_mul(float* __restrict__ a, const float* __restrict__ g) {
    const size_t i = (size_t)blockIdx.x * 256 + threadIdx.x;
    const float v = a[i];
    a[i] = (v / (1.0f + __expf(-v))) * g[i];
}

// ---------------------------------------------------------------------------
// Output projection: out[m,c] = h[m,:] . W_out[:,c] + b_out[c]  (N=16)
// ---------------------------------------------------------------------------
__global__ void __launch_bounds__(256)
out_proj(const float* __restrict__ hp, const float* __restrict__ Wout,
         const float* __restrict__ bout, float* __restrict__ outp) {
    const int idx = blockIdx.x * 256 + threadIdx.x;  // < 8192*16
    const int m = idx >> 4, c = idx & 15;
    float acc = bout[c];
    const float* hr = hp + (size_t)m * 512;
    for (int k2 = 0; k2 < 512; k2++) acc += hr[k2] * Wout[k2 * 16 + c];
    outp[idx] = acc;
}

// ---------------------------------------------------------------------------
// Host launcher
// ---------------------------------------------------------------------------
extern "C" void kernel_launch(void* const* d_in, const int* in_sizes, int n_in,
                              void* d_out, int out_size, void* d_ws, size_t ws_size,
                              hipStream_t stream) {
    (void)in_sizes; (void)n_in; (void)out_size; (void)ws_size;

    const float* x        = (const float*)d_in[0];
    const int*   tok_mask = (const int*)d_in[1];
    const float* W_in     = (const float*)d_in[2];
    const float* b_in     = (const float*)d_in[3];
    const float* ln1_g    = (const float*)d_in[4];
    const float* ln1_b    = (const float*)d_in[5];
    const float* Wq       = (const float*)d_in[6];
    const float* Wk       = (const float*)d_in[7];
    const float* Wv       = (const float*)d_in[8];
    const float* Wo       = (const float*)d_in[9];
    const float* ln2_g    = (const float*)d_in[10];
    const float* ln2_b    = (const float*)d_in[11];
    const float* W1       = (const float*)d_in[12];
    const float* W2       = (const float*)d_in[13];
    const float* W3       = (const float*)d_in[14];
    const float* W_out    = (const float*)d_in[15];
    const float* b_out    = (const float*)d_in[16];
    // d_in[17], d_in[18] = H, W (hardcoded 32x32 -> L=1024)

    const int NL = 12, E = 512, FFNF = 2048;

    char* ws = (char*)d_ws;
    float*    h    = (float*)(ws + 0);           //  16 MiB: 8192x512 f32
    float*    xn   = (float*)(ws + 16777216);    //  16 MiB
    _Float16* q16  = (_Float16*)(ws + 33554432); //   8 MiB: 8192x512 f16
    _Float16* k16  = (_Float16*)(ws + 41943040); //   8 MiB
    _Float16* v16p = (_Float16*)(ws + 50331648); //   8 MiB
    float*    attn = (float*)(ws + 58720256);    //  16 MiB
    float*    ff1  = (float*)(ws + 75497472);    //  64 MiB: 8192x2048 f32
    float*    ff2  = (float*)(ws + 142606336);   //  64 MiB  (end: 200 MiB)

    input_proj<<<16384, 256, 0, stream>>>(x, W_in, b_in, h);

    const dim3 g512(512 / 128, 8192 / 128);
    const dim3 g2048(2048 / 128, 8192 / 128);

    for (int lyr = 0; lyr < NL; lyr++) {
        ln_kernel<<<8192, 256, 0, stream>>>(h, ln1_g + lyr * E, ln1_b + lyr * E, xn);
        gemm_f16wmma<2><<<g512, 256, 0, stream>>>(xn, Wq + (size_t)lyr * E * E, q16, 8192, 512, 512);
        gemm_f16wmma<2><<<g512, 256, 0, stream>>>(xn, Wk + (size_t)lyr * E * E, k16, 8192, 512, 512);
        gemm_f16wmma<2><<<g512, 256, 0, stream>>>(xn, Wv + (size_t)lyr * E * E, v16p, 8192, 512, 512);
        flash_attn<<<dim3(16, 8, 8), 128, 0, stream>>>(q16, k16, v16p, tok_mask, attn);
        gemm_f16wmma<1><<<g512, 256, 0, stream>>>(attn, Wo + (size_t)lyr * E * E, h, 8192, 512, 512);
        ln_kernel<<<8192, 256, 0, stream>>>(h, ln2_g + lyr * E, ln2_b + lyr * E, xn);
        gemm_f16wmma<0><<<g2048, 256, 0, stream>>>(xn, W1 + (size_t)lyr * E * FFNF, ff1, 8192, 2048, 512);
        gemm_f16wmma<0><<<g2048, 256, 0, stream>>>(xn, W2 + (size_t)lyr * E * FFNF, ff2, 8192, 2048, 512);
        silu_mul<<<65536, 256, 0, stream>>>(ff1, ff2);
        gemm_f16wmma<1><<<g512, 256, 0, stream>>>(ff1, W3 + (size_t)lyr * FFNF * E, h, 8192, 512, 2048);
    }

    out_proj<<<512, 256, 0, stream>>>(h, W_out, b_out, (float*)d_out);
}
